// AIGGenerator_31533649887522
// MI455X (gfx1250) — compile-verified, matrix operand order checked by reading, not run on hardware
//
#include <hip/hip_runtime.h>
#include <hip/hip_bf16.h>

#define HDIM  128
#define KCAND 128
#define TSTR  136   // padded LDS row stride in halfwords (avoid bank conflicts)

typedef __attribute__((ext_vector_type(16))) _Float16 v16h;
typedef __attribute__((ext_vector_type(8)))  float    v8f;

// ---------------- WMMA fragment loaders (16x16x32 f16, wave32) ----------------
// A 16x32 f16: lane l -> row m = l&15; elems 0..7 : k = kbase..kbase+7
//                                      elems 8..15: k = kbase+16..kbase+23
// with kbase = kblk*32 + (l>>4)*8.  B 32x16 is symmetric with n = l&15.
__device__ inline v16h load_frag_lds(const _Float16* buf, int row, int kbase) {
  v16h r;
  const _Float16* p = buf + row * TSTR + kbase;
#pragma unroll
  for (int i = 0; i < 8; ++i) { r[i] = p[i]; r[8 + i] = p[16 + i]; }
  return r;
}

__device__ inline v16h load_frag_w(const _Float16* W, int row, int kbase) {
  v16h r;
  const _Float16* p = W + row * HDIM + kbase;
#pragma unroll
  for (int i = 0; i < 8; ++i) { r[i] = p[i]; r[8 + i] = p[16 + i]; }
  return r;
}

__device__ inline float dot4(const float4& a, const float4& b) {
  return a.x * b.x + a.y * b.y + a.z * b.z + a.w * b.w;
}

// ---------------- K0: fold z into biases, convert weights to f16 ----------------
__global__ __launch_bounds__(256) void aig_prep(
    const float* __restrict__ W_enc2, const float* __restrict__ W_proj,
    const float* __restrict__ W_src,  const float* __restrict__ W_tgt,
    const float* __restrict__ b_proj, const float* __restrict__ z,
    const float* __restrict__ W_inv1, const float* __restrict__ b_inv1,
    _Float16* __restrict__ w2h, _Float16* __restrict__ wph,
    _Float16* __restrict__ wsh, _Float16* __restrict__ wth,
    float* __restrict__ cproj, float* __restrict__ cinv)
{
  const int t = threadIdx.x;
  for (int i = t; i < HDIM * HDIM; i += 256) {
    int n = i >> 7, k = i & (HDIM - 1);
    w2h[i] = (_Float16)W_enc2[i];
    wsh[i] = (_Float16)W_src[i];
    wth[i] = (_Float16)W_tgt[i];
    wph[i] = (_Float16)W_proj[n * 256 + k];           // first 128 cols of W_proj
  }
  if (t < HDIM) {
    float s = b_proj[t];
    for (int q = 0; q < HDIM; ++q) s += W_proj[t * 256 + 128 + q] * z[q];
    cproj[t] = s;
    float s2 = b_inv1[t];
    for (int q = 0; q < HDIM; ++q) s2 += W_inv1[t * 384 + 256 + q] * z[q];
    cinv[t] = s2;
  }
}

// ---------------- K1: fused WMMA chain  a -> h0 -> h -> hs/ht ----------------
__global__ __launch_bounds__(256) void aig_gemm_chain(
    const float* __restrict__ x, const float* __restrict__ W_enc1,
    const float* __restrict__ b_enc1, const float* __restrict__ b_enc2,
    const _Float16* __restrict__ w2h, const _Float16* __restrict__ wph,
    const _Float16* __restrict__ wsh, const _Float16* __restrict__ wth,
    const float* __restrict__ cproj,
    float* __restrict__ hOut, float* __restrict__ hsOut, float* __restrict__ htOut,
    int N)
{
  __shared__ _Float16 aBuf[8][16 * TSTR];
  __shared__ _Float16 hBuf[8][16 * TSTR];
  __shared__ float sW1a[HDIM], sW1b[HDIM], sB1[HDIM], sB2[HDIM], sCp[HDIM];

  const int t = threadIdx.x, lane = t & 31, w = t >> 5;
  if (t < HDIM) {
    sW1a[t] = W_enc1[2 * t]; sW1b[t] = W_enc1[2 * t + 1];
    sB1[t] = b_enc1[t]; sB2[t] = b_enc2[t]; sCp[t] = cproj[t];
  }
  __syncthreads();

  const int tileBase = (blockIdx.x * 8 + w) * 16;
  _Float16* A  = aBuf[w];
  _Float16* Hb = hBuf[w];

  // stage 0: first encoder layer (K=2, VALU), emit f16 activation tile
  for (int idx = lane; idx < 16 * HDIM; idx += 32) {
    int m = idx >> 7, j = idx & (HDIM - 1);
    int node = tileBase + m;
    float x0 = 0.f, x1 = 0.f;
    if (node < N) { x0 = x[2 * node]; x1 = x[2 * node + 1]; }
    float a = sW1a[j] * x0 + sW1b[j] * x1 + sB1[j];
    A[m * TSTR + j] = (_Float16)(a > 0.f ? a : 0.f);
  }
  __syncthreads();

  const int nl = lane & 15, hi = lane >> 4;

  // stage 1: h0 = a @ W_enc2^T + b2   (no relu)
  for (int nblk = 0; nblk < 8; ++nblk) {
    v8f c;
#pragma unroll
    for (int i = 0; i < 8; ++i) c[i] = 0.f;
#pragma unroll
    for (int kblk = 0; kblk < 4; ++kblk) {
      v16h af = load_frag_lds(A, nl, kblk * 32 + hi * 8);
      v16h bf = load_frag_w(w2h, nblk * 16 + nl, kblk * 32 + hi * 8);
      c = __builtin_amdgcn_wmma_f32_16x16x32_f16(false, af, false, bf, (short)0, c, false, false);
    }
    int n = nblk * 16 + nl;
#pragma unroll
    for (int r = 0; r < 8; ++r) {
      int m = r + 8 * hi;
      Hb[m * TSTR + n] = (_Float16)(c[r] + sB2[n]);
    }
  }
  __syncthreads();

  // stage 2: h = relu(h0 @ Wp1^T + cproj)   (z folded into cproj)
  for (int nblk = 0; nblk < 8; ++nblk) {
    v8f c;
#pragma unroll
    for (int i = 0; i < 8; ++i) c[i] = 0.f;
#pragma unroll
    for (int kblk = 0; kblk < 4; ++kblk) {
      v16h af = load_frag_lds(Hb, nl, kblk * 32 + hi * 8);
      v16h bf = load_frag_w(wph, nblk * 16 + nl, kblk * 32 + hi * 8);
      c = __builtin_amdgcn_wmma_f32_16x16x32_f16(false, af, false, bf, (short)0, c, false, false);
    }
    int n = nblk * 16 + nl;
#pragma unroll
    for (int r = 0; r < 8; ++r) {
      int m = r + 8 * hi;
      float v = c[r] + sCp[n];
      v = v > 0.f ? v : 0.f;
      A[m * TSTR + n] = (_Float16)v;
      int node = tileBase + m;
      if (node < N) hOut[(size_t)node * HDIM + n] = v;
    }
  }
  __syncthreads();

  // stage 3: hs = h @ W_src^T
  for (int nblk = 0; nblk < 8; ++nblk) {
    v8f c;
#pragma unroll
    for (int i = 0; i < 8; ++i) c[i] = 0.f;
#pragma unroll
    for (int kblk = 0; kblk < 4; ++kblk) {
      v16h af = load_frag_lds(A, nl, kblk * 32 + hi * 8);
      v16h bf = load_frag_w(wsh, nblk * 16 + nl, kblk * 32 + hi * 8);
      c = __builtin_amdgcn_wmma_f32_16x16x32_f16(false, af, false, bf, (short)0, c, false, false);
    }
    int n = nblk * 16 + nl;
#pragma unroll
    for (int r = 0; r < 8; ++r) {
      int m = r + 8 * hi, node = tileBase + m;
      if (node < N) hsOut[(size_t)node * HDIM + n] = c[r];
    }
  }

  // stage 4: ht = h @ W_tgt^T
  for (int nblk = 0; nblk < 8; ++nblk) {
    v8f c;
#pragma unroll
    for (int i = 0; i < 8; ++i) c[i] = 0.f;
#pragma unroll
    for (int kblk = 0; kblk < 4; ++kblk) {
      v16h af = load_frag_lds(A, nl, kblk * 32 + hi * 8);
      v16h bf = load_frag_w(wth, nblk * 16 + nl, kblk * 32 + hi * 8);
      c = __builtin_amdgcn_wmma_f32_16x16x32_f16(false, af, false, bf, (short)0, c, false, false);
    }
    int n = nblk * 16 + nl;
#pragma unroll
    for (int r = 0; r < 8; ++r) {
      int m = r + 8 * hi, node = tileBase + m;
      if (node < N) htOut[(size_t)node * HDIM + n] = c[r];
    }
  }
}

// ---------------- K2: gather-scoring + top2 + inv MLP (one wave per node) ----------------
__global__ __launch_bounds__(256) void aig_score_top2_inv(
    const float* __restrict__ hs, const float* __restrict__ ht, const float* __restrict__ h,
    const int* __restrict__ cand_idx, const int* __restrict__ cand_mask,
    const int* __restrict__ node_type,
    const float* __restrict__ W_inv1, const float* __restrict__ cinv,
    const float* __restrict__ w_inv2, const float* __restrict__ b_inv2,
    float* __restrict__ out, int N)
{
  const int lane = threadIdx.x & 31;
  const int n = blockIdx.x * 8 + (threadIdx.x >> 5);
  if (n >= N) return;

  // ---- candidate scoring: score[k] = dot(hs[cand[k]], ht[n]) ----
  float4 tfrag = ((const float4*)(ht + (size_t)n * HDIM))[lane];  // ht[4*lane..+3]
  int cidx[4]; int cm[4]; const float4* hsrow[4];
#pragma unroll
  for (int j = 0; j < 4; ++j) {
    int k = 32 * j + lane;
    cidx[j]  = cand_idx[(size_t)n * KCAND + k];
    cm[j]    = cand_mask[(size_t)n * KCAND + k];
    hsrow[j] = (const float4*)(hs + (size_t)cidx[j] * HDIM);
  }
  float acc[4] = {0.f, 0.f, 0.f, 0.f};
  for (int kk = 0; kk < 32; ++kk) {
    float4 tv;
    tv.x = __shfl(tfrag.x, kk, 32); tv.y = __shfl(tfrag.y, kk, 32);
    tv.z = __shfl(tfrag.z, kk, 32); tv.w = __shfl(tfrag.w, kk, 32);
#pragma unroll
    for (int j = 0; j < 4; ++j) {
      float4 hv = hsrow[j][kk];
      acc[j] += dot4(tv, hv);
    }
  }

  // ---- masked wave-wide top-2 (value, slot-index) ----
  float v1 = -3e38f, v2 = -3e38f; int i1 = 0, i2 = 0;
#pragma unroll
  for (int j = 0; j < 4; ++j) {
    float s = cm[j] ? acc[j] : -1e30f;
    int k = 32 * j + lane;
    if (s > v1) { v2 = v1; i2 = i1; v1 = s; i1 = k; }
    else if (s > v2) { v2 = s; i2 = k; }
  }
#pragma unroll
  for (int off = 16; off >= 1; off >>= 1) {
    float ov1 = __shfl_xor(v1, off, 32); int oi1 = __shfl_xor(i1, off, 32);
    float ov2 = __shfl_xor(v2, off, 32); int oi2 = __shfl_xor(i2, off, 32);
    if (ov1 > v1) { v2 = v1; i2 = i1; v1 = ov1; i1 = oi1; }
    else if (ov1 > v2) { v2 = ov1; i2 = oi1; }
    if (ov2 > v1) { v2 = v1; i2 = i1; v1 = ov2; i1 = oi2; }
    else if (ov2 > v2) { v2 = ov2; i2 = oi2; }
  }

  const int ntype = node_type[n];
  const int kpn = (ntype == 0) ? 0 : ((ntype == 2) ? 2 : 1);
  float4 hb4 = ((const float4*)(h + (size_t)n * HDIM))[lane];

  float logitE[2]; int srcE[2]; int validE[2];
  const int ksel0 = i1, ksel1 = i2;
#pragma unroll
  for (int e = 0; e < 2; ++e) {
    int k = (e == 0) ? ksel0 : ksel1;
    int s  = cand_idx[(size_t)n * KCAND + k];
    int mk = cand_mask[(size_t)n * KCAND + k];
    srcE[e] = s;
    validE[e] = (e < kpn) && (mk != 0);

    // inv MLP: g = relu(Wa@hu + Wb@h[n] + cinv); logit = w2.g + b
    float4 hu4 = ((const float4*)(h + (size_t)s * HDIM))[lane];
    float g[4];
    const float4* wr[4];
#pragma unroll
    for (int q = 0; q < 4; ++q) {
      int jj = lane + 32 * q;
      g[q] = cinv[jj];
      wr[q] = (const float4*)(W_inv1 + (size_t)jj * 384);
    }
    for (int kk = 0; kk < 32; ++kk) {
      float4 bu, bt;
      bu.x = __shfl(hu4.x, kk, 32); bu.y = __shfl(hu4.y, kk, 32);
      bu.z = __shfl(hu4.z, kk, 32); bu.w = __shfl(hu4.w, kk, 32);
      bt.x = __shfl(hb4.x, kk, 32); bt.y = __shfl(hb4.y, kk, 32);
      bt.z = __shfl(hb4.z, kk, 32); bt.w = __shfl(hb4.w, kk, 32);
#pragma unroll
      for (int q = 0; q < 4; ++q) {
        float4 wu = wr[q][kk];
        float4 wt = wr[q][32 + kk];
        g[q] += dot4(bu, wu) + dot4(bt, wt);
      }
    }
    float part = 0.f;
#pragma unroll
    for (int q = 0; q < 4; ++q) {
      int jj = lane + 32 * q;
      float gr = g[q] > 0.f ? g[q] : 0.f;
      part += gr * w_inv2[jj];
    }
#pragma unroll
    for (int off = 16; off >= 1; off >>= 1) part += __shfl_xor(part, off, 32);
    logitE[e] = part + b_inv2[0];
  }

  if (lane == 0) {
    float* logits_o = out;
    float* src_o    = out + 2 * (size_t)N;
    float* valid_o  = out + 4 * (size_t)N;
    float* attr_o   = out + 6 * (size_t)N;
#pragma unroll
    for (int e = 0; e < 2; ++e) {
      logits_o[2 * n + e] = validE[e] ? logitE[e] : 0.f;
      src_o[2 * n + e]    = (float)srcE[e];
      valid_o[2 * n + e]  = validE[e] ? 1.f : 0.f;
      attr_o[2 * n + e]   = (validE[e] && (logitE[e] > 0.f)) ? 1.f : 0.f;
    }
  }
}

extern "C" void kernel_launch(void* const* d_in, const int* in_sizes, int n_in,
                              void* d_out, int out_size, void* d_ws, size_t ws_size,
                              hipStream_t stream) {
  (void)n_in; (void)out_size; (void)ws_size;
  const float* x      = (const float*)d_in[0];
  const float* z      = (const float*)d_in[1];
  const float* W_enc1 = (const float*)d_in[2];
  const float* b_enc1 = (const float*)d_in[3];
  const float* W_enc2 = (const float*)d_in[4];
  const float* b_enc2 = (const float*)d_in[5];
  const float* W_proj = (const float*)d_in[6];
  const float* b_proj = (const float*)d_in[7];
  const float* W_src  = (const float*)d_in[8];
  const float* W_tgt  = (const float*)d_in[9];
  const float* W_inv1 = (const float*)d_in[10];
  const float* b_inv1 = (const float*)d_in[11];
  const float* w_inv2 = (const float*)d_in[12];
  const float* b_inv2 = (const float*)d_in[13];
  const int* node_type = (const int*)d_in[14];
  const int* cand_idx  = (const int*)d_in[15];
  const int* cand_mask = (const int*)d_in[16];

  const int N = in_sizes[0] / 2;   // x is [N, DIN=2]
  float* out = (float*)d_out;

  // workspace carve (all 256B-aligned chunks)
  char* ws = (char*)d_ws;
  size_t off = 0;
  auto carve = [&](size_t bytes) -> char* {
    char* p = ws + off;
    off += (bytes + 255) & ~(size_t)255;
    return p;
  };
  _Float16* w2h = (_Float16*)carve((size_t)HDIM * HDIM * 2);
  _Float16* wph = (_Float16*)carve((size_t)HDIM * HDIM * 2);
  _Float16* wsh = (_Float16*)carve((size_t)HDIM * HDIM * 2);
  _Float16* wth = (_Float16*)carve((size_t)HDIM * HDIM * 2);
  float* cproj  = (float*)carve(HDIM * 4);
  float* cinv   = (float*)carve(HDIM * 4);
  float* hbuf   = (float*)carve((size_t)N * HDIM * 4);
  float* hsbuf  = (float*)carve((size_t)N * HDIM * 4);
  float* htbuf  = (float*)carve((size_t)N * HDIM * 4);

  aig_prep<<<1, 256, 0, stream>>>(W_enc2, W_proj, W_src, W_tgt, b_proj, z,
                                  W_inv1, b_inv1, w2h, wph, wsh, wth, cproj, cinv);

  int blocks1 = (N + 127) / 128;   // 8 waves/block, 16 nodes/wave
  aig_gemm_chain<<<blocks1, 256, 0, stream>>>(x, W_enc1, b_enc1, b_enc2,
                                              w2h, wph, wsh, wth, cproj,
                                              hbuf, hsbuf, htbuf, N);

  int blocks2 = (N + 7) / 8;       // one wave per node
  aig_score_top2_inv<<<blocks2, 256, 0, stream>>>(hsbuf, htbuf, hbuf,
                                                  cand_idx, cand_mask, node_type,
                                                  W_inv1, cinv, w_inv2, b_inv2,
                                                  out, N);
}
